// ROPEAttentionGroupSequenceIngroupLayer_10780367913270
// MI455X (gfx1250) — matrix operand order repaired and checked by reading, not compiled, not run on hardware
//
#include <hip/hip_runtime.h>
#include <hip/hip_bf16.h>
#include <math.h>

// ---------------------------------------------------------------------------
// ROPE attention for MI455X (gfx1250, wave32, WMMA).
// Compute-bound (~330 FLOP/byte): V_WMMA_F32_16X16X32_BF16 with a compensated
// hi/lo bf16 split (3 WMMAs per tile, f32 accumulate) for fp32-grade accuracy
// at ~2.7x the K-throughput of V_WMMA_F32_16X16X4_F32.
// All LDS traffic vectorized: fragments via 2x ds_load_b128, staging via
// ds_store_b64. GEMM K-loop is software-pipelined (register-staged tiles).
// ---------------------------------------------------------------------------

typedef __attribute__((ext_vector_type(16))) __bf16 v16bf;
typedef __attribute__((ext_vector_type(8)))  __bf16 v8bf;
typedef __attribute__((ext_vector_type(4)))  __bf16 v4bf;
typedef __attribute__((ext_vector_type(2)))  __bf16 v2bf;
typedef __attribute__((ext_vector_type(8)))  float  v8f;

#define DM   1024   // d_model
#define DQK  1024   // n_heads * d_keys
#define NB   64     // batch
#define LSEQ 128    // L == S
#define NH   16
#define DK   64
#define BL   (NB * LSEQ)   // 8192 rows

// ---------------------------------------------------------------------------
// helpers
// ---------------------------------------------------------------------------

// split 4 fp32 into bf16 hi/lo quads and store as two b64 LDS writes
static __device__ inline void split4_store(float x0, float x1, float x2, float x3,
                                           __bf16* hp, __bf16* lp) {
  v4bf h, l;
  __bf16 t;
  t = (__bf16)x0; h[0] = t; l[0] = (__bf16)(x0 - (float)t);
  t = (__bf16)x1; h[1] = t; l[1] = (__bf16)(x1 - (float)t);
  t = (__bf16)x2; h[2] = t; l[2] = (__bf16)(x2 - (float)t);
  t = (__bf16)x3; h[3] = t; l[3] = (__bf16)(x3 - (float)t);
  *(v4bf*)hp = h;
  *(v4bf*)lp = l;
}

// A fragment (16x32 bf16, MxK): lanes 0-15 hold M=lane, K={0..7,16..23};
// lanes 16-31 hold M=lane-16, K={8..15,24..31}. rowp -> start of 32-wide K seg.
// Two 16B-aligned v8bf loads -> 2x ds_load_b128.
static __device__ inline v16bf load_a_frag(const __bf16* rowp, int lane) {
  const int k0 = (lane & 16) ? 8 : 0;
  v8bf a0 = *(const v8bf*)(rowp + k0);
  v8bf a1 = *(const v8bf*)(rowp + k0 + 16);
  return __builtin_shufflevector(a0, a1, 0, 1, 2, 3, 4, 5, 6, 7,
                                         8, 9, 10, 11, 12, 13, 14, 15);
}

// B fragment (32x16 bf16, KxN): lanes 0-15 hold N=lane, K={0..7,16..23};
// lanes 16-31 hold N=lane-16, K={8..15,24..31}. rowp -> row N of K-major tile.
static __device__ inline v16bf load_b_frag(const __bf16* rowp, int lane) {
  const int k0 = (lane & 16) ? 8 : 0;
  v8bf b0 = *(const v8bf*)(rowp + k0);
  v8bf b1 = *(const v8bf*)(rowp + k0 + 16);
  return __builtin_shufflevector(b0, b1, 0, 1, 2, 3, 4, 5, 6, 7,
                                         8, 9, 10, 11, 12, 13, 14, 15);
}

// compensated product: acc += ah*bh + ah*bl + al*bh   (f32 accumulate)
static __device__ inline v8f wmma_comp(v16bf ah, v16bf al, v16bf bh, v16bf bl, v8f acc) {
  acc = __builtin_amdgcn_wmma_f32_16x16x32_bf16(false, al, false, bh, (short)0, acc, false, false);
  acc = __builtin_amdgcn_wmma_f32_16x16x32_bf16(false, ah, false, bl, (short)0, acc, false, false);
  acc = __builtin_amdgcn_wmma_f32_16x16x32_bf16(false, ah, false, bh, (short)0, acc, false, false);
  return acc;
}

// ---------------------------------------------------------------------------
// GEMM: C[M,N] = A[M,K] @ W[K,N] + bias[N]   (fp32 in/out, bf16x2 WMMA core)
// 128x128 tile per workgroup, 8 waves in 2(M) x 4(N); each wave 4x2 16x16
// tiles. K-loop software-pipelined: next tile staged in VGPRs during compute.
// ---------------------------------------------------------------------------
#define GA_STR 40  // LDS row stride in bf16 elems (80B: 16B aligned, conflict-free)

__global__ __launch_bounds__(256) void gemm_bias_kernel(
    const float* __restrict__ A, const float* __restrict__ W,
    const float* __restrict__ bias, float* __restrict__ C,
    int M, int N, int K) {
  __shared__ __bf16 AsH[128][GA_STR];
  __shared__ __bf16 AsL[128][GA_STR];
  __shared__ __bf16 BsH[128][GA_STR];   // transposed: [n][k]
  __shared__ __bf16 BsL[128][GA_STR];

  const int tid  = threadIdx.x;
  const int lane = tid & 31;
  const int wave = tid >> 5;
  const int waveM = wave & 1;      // 0..1 -> 64-row half
  const int waveN = wave >> 1;     // 0..3 -> 32-col quarter
  const int m0 = blockIdx.y * 128;
  const int n0 = blockIdx.x * 128;

  // staging geometry (per thread, 4 units each for A and B)
  int arow[4], acol[4], bn[4], bk[4];
#pragma unroll
  for (int i = 0; i < 4; ++i) {
    int ia = tid + i * 256;            // 0..1023
    arow[i] = ia >> 3;                 // 0..127
    acol[i] = (ia & 7) * 4;            // 0..28
    bn[i]   = ia & 127;                // 0..127 (consecutive lanes -> coalesced)
    bk[i]   = (ia >> 7) * 4;           // 0..28
  }

  float4 ra[4];
  float  rb[4][4];

  auto load_tiles = [&](int k0) {
#pragma unroll
    for (int i = 0; i < 4; ++i)
      ra[i] = *(const float4*)(A + (size_t)(m0 + arow[i]) * K + k0 + acol[i]);
#pragma unroll
    for (int i = 0; i < 4; ++i)
#pragma unroll
      for (int j = 0; j < 4; ++j)
        rb[i][j] = W[(size_t)(k0 + bk[i] + j) * N + n0 + bn[i]];
  };
  auto store_tiles = [&]() {
#pragma unroll
    for (int i = 0; i < 4; ++i) {
      split4_store(ra[i].x, ra[i].y, ra[i].z, ra[i].w,
                   &AsH[arow[i]][acol[i]], &AsL[arow[i]][acol[i]]);
      split4_store(rb[i][0], rb[i][1], rb[i][2], rb[i][3],
                   &BsH[bn[i]][bk[i]], &BsL[bn[i]][bk[i]]);
    }
  };

  v8f acc[4][2] = {};
  load_tiles(0);

  for (int k0 = 0; k0 < K; k0 += 32) {
    store_tiles();
    __syncthreads();
    if (k0 + 32 < K) load_tiles(k0 + 32);   // overlap global latency with WMMA

#pragma unroll
    for (int mt = 0; mt < 4; ++mt) {
      const int ar = waveM * 64 + mt * 16 + (lane & 15);
      v16bf ah = load_a_frag(&AsH[ar][0], lane);
      v16bf al = load_a_frag(&AsL[ar][0], lane);
#pragma unroll
      for (int nt = 0; nt < 2; ++nt) {
        const int br = waveN * 32 + nt * 16 + (lane & 15);
        v16bf bh = load_b_frag(&BsH[br][0], lane);
        v16bf bl = load_b_frag(&BsL[br][0], lane);
        acc[mt][nt] = wmma_comp(ah, al, bh, bl, acc[mt][nt]);
      }
    }
    __syncthreads();
  }

  // epilogue: C layout VGPR r -> M = r + (lane>=16 ? 8 : 0), N = lane%16
#pragma unroll
  for (int mt = 0; mt < 4; ++mt) {
#pragma unroll
    for (int nt = 0; nt < 2; ++nt) {
      const int n = n0 + waveN * 32 + nt * 16 + (lane & 15);
      const float bvv = bias[n];
      const int mbase = m0 + waveM * 64 + mt * 16 + ((lane & 16) ? 8 : 0);
#pragma unroll
      for (int r = 0; r < 8; ++r)
        C[(size_t)(mbase + r) * N + n] = acc[mt][nt][r] + bvv;
    }
  }
}

// ---------------------------------------------------------------------------
// RoPE (ingroup then seq) applied in-place to q and k workspace buffers.
// Layout [B*L, H*E]; tables [128][64] with sin in cols 0..31, cos in 32..63.
// ---------------------------------------------------------------------------
__global__ __launch_bounds__(256) void rope_kernel(
    float* __restrict__ q, float* __restrict__ k,
    const float* __restrict__ ingroup, const float* __restrict__ seq) {
  const size_t idx = (size_t)blockIdx.x * blockDim.x + threadIdx.x;  // 8192*512
  const int pair = (int)(idx & 511);       // h*32 + i
  const int row  = (int)(idx >> 9);        // b*128 + l
  const int l = row & 127;
  const int i = pair & 31;
  const int h = pair >> 5;
  const size_t off = (size_t)row * DQK + h * DK + 2 * i;

  const float s1 = ingroup[l * DK + i], c1 = ingroup[l * DK + 32 + i];
  const float s2 = seq[l * DK + i],     c2 = seq[l * DK + 32 + i];

  {
    float2 x = *(const float2*)(q + off);
    float y0 = x.x * c1 - x.y * s1, y1 = x.y * c1 + x.x * s1;
    float2 z; z.x = y0 * c2 - y1 * s2; z.y = y1 * c2 + y0 * s2;
    *(float2*)(q + off) = z;
  }
  {
    float2 x = *(const float2*)(k + off);
    float y0 = x.x * c1 - x.y * s1, y1 = x.y * c1 + x.x * s1;
    float2 z; z.x = y0 * c2 - y1 * s2; z.y = y1 * c2 + y0 * s2;
    *(float2*)(k + off) = z;
  }
}

// ---------------------------------------------------------------------------
// Attention: one (b,h) per block. scores -> softmax (attn output) -> AV.
// Static LDS, phase-aliased regions:
//   phase1: qs/ks hi+lo (0..73727) + sc f32 (73728..141311)
//   phase2: pt hi+lo (0..69631) + sc
//   phase3: pt + vt hi+lo (73728..108543)
// ---------------------------------------------------------------------------
#define QS_STR 72
#define PT_STR 136
#define VT_STR 136
#define SC_STR 132

#define OFF_QS_H 0
#define OFF_QS_L 18432
#define OFF_KS_H 36864
#define OFF_KS_L 55296
#define OFF_PT_H 0
#define OFF_PT_L 34816
#define OFF_SC   73728
#define OFF_VT_H 73728
#define OFF_VT_L 91136
#define OFF_MX   141312
#define OFF_RS   141824
#define SMEM_BYTES 142336

__global__ __launch_bounds__(256) void attn_kernel(
    const float* __restrict__ qbuf, const float* __restrict__ kbuf,
    const float* __restrict__ vbuf, const unsigned char* __restrict__ mask,
    float* __restrict__ attn_out, float* __restrict__ ctx) {
  __shared__ __align__(16) unsigned char smem[SMEM_BYTES];

  __bf16 (*qsH)[QS_STR] = (__bf16(*)[QS_STR])(smem + OFF_QS_H);
  __bf16 (*qsL)[QS_STR] = (__bf16(*)[QS_STR])(smem + OFF_QS_L);
  __bf16 (*ksH)[QS_STR] = (__bf16(*)[QS_STR])(smem + OFF_KS_H);
  __bf16 (*ksL)[QS_STR] = (__bf16(*)[QS_STR])(smem + OFF_KS_L);
  float* sc = (float*)(smem + OFF_SC);
  float* mx = (float*)(smem + OFF_MX);
  float* rs = (float*)(smem + OFF_RS);

  const int tid = threadIdx.x;
  const int lane = tid & 31;
  const int wave = tid >> 5;
  const int b = blockIdx.x >> 4;
  const int h = blockIdx.x & 15;
  const size_t rowbase = (size_t)b * LSEQ * DQK + (size_t)h * DK;

  // ---- stage q, k (128 x 64 fp32 -> bf16 hi/lo, b64 LDS stores) ----
#pragma unroll
  for (int i = 0; i < 8; ++i) {
    int idx = tid + i * 256;               // 0..2047 float4 units
    int r = idx >> 4, c = (idx & 15) * 4;
    float4 fq = *(const float4*)(qbuf + rowbase + (size_t)r * DQK + c);
    split4_store(fq.x, fq.y, fq.z, fq.w, &qsH[r][c], &qsL[r][c]);
    float4 fk = *(const float4*)(kbuf + rowbase + (size_t)r * DQK + c);
    split4_store(fk.x, fk.y, fk.z, fk.w, &ksH[r][c], &ksL[r][c]);
  }
  __syncthreads();

  // ---- scores = q @ k^T * scale  (M=128, N=128, K=64) ----
  {
    const int waveM = wave & 1, waveN = wave >> 1;
    v8f acc[4][2] = {};
#pragma unroll
    for (int ks = 0; ks < 64; ks += 32) {
#pragma unroll
      for (int mt = 0; mt < 4; ++mt) {
        const int ar = waveM * 64 + mt * 16 + (lane & 15);
        v16bf ah = load_a_frag(&qsH[ar][ks], lane);
        v16bf al = load_a_frag(&qsL[ar][ks], lane);
#pragma unroll
        for (int nt = 0; nt < 2; ++nt) {
          const int br = waveN * 32 + nt * 16 + (lane & 15);
          v16bf bh = load_b_frag(&ksH[br][ks], lane);
          v16bf bl = load_b_frag(&ksL[br][ks], lane);
          acc[mt][nt] = wmma_comp(ah, al, bh, bl, acc[mt][nt]);
        }
      }
    }
    const float scale = 0.125f;   // 1/sqrt(64)
#pragma unroll
    for (int mt = 0; mt < 4; ++mt)
#pragma unroll
      for (int nt = 0; nt < 2; ++nt) {
        const int n = waveN * 32 + nt * 16 + (lane & 15);
        const int mbase = waveM * 64 + mt * 16 + ((lane & 16) ? 8 : 0);
#pragma unroll
        for (int r = 0; r < 8; ++r)
          sc[(mbase + r) * SC_STR + n] = acc[mt][nt][r] * scale;
      }
  }
  __syncthreads();

  // ---- softmax pass 1: mask, row max, row sum ----
  if (tid < 128) {
    const unsigned char* mrow = mask + ((size_t)b * LSEQ + tid) * LSEQ;
    float m = -INFINITY;
    for (int s = 0; s < 128; ++s) {
      float v = sc[tid * SC_STR + s];
      if (mrow[s]) { v = -INFINITY; sc[tid * SC_STR + s] = v; }
      m = fmaxf(m, v);
    }
    float sum = 0.f;
    for (int s = 0; s < 128; ++s) sum += __expf(sc[tid * SC_STR + s] - m);
    mx[tid] = m;
    rs[tid] = 1.0f / sum;
  }
  __syncthreads();

  // ---- softmax pass 2: probs -> attn output (float2) + bf16 hi/lo tile ----
  __bf16 (*ptH)[PT_STR] = (__bf16(*)[PT_STR])(smem + OFF_PT_H);  // aliases qs/ks (dead)
  __bf16 (*ptL)[PT_STR] = (__bf16(*)[PT_STR])(smem + OFF_PT_L);
  float* attn_row = attn_out + (size_t)blockIdx.x * LSEQ * LSEQ;
  for (int idx = tid; idx < (LSEQ * LSEQ) / 2; idx += 256) {
    int r = idx >> 6, s = (idx & 63) * 2;
    float rm = mx[r], rr = rs[r];
    float p0 = __expf(sc[r * SC_STR + s] - rm) * rr;
    float p1 = __expf(sc[r * SC_STR + s + 1] - rm) * rr;
    float2 pp; pp.x = p0; pp.y = p1;
    *(float2*)(attn_row + r * LSEQ + s) = pp;
    v2bf hh, ll;
    __bf16 t;
    t = (__bf16)p0; hh[0] = t; ll[0] = (__bf16)(p0 - (float)t);
    t = (__bf16)p1; hh[1] = t; ll[1] = (__bf16)(p1 - (float)t);
    *(v2bf*)&ptH[r][s] = hh;
    *(v2bf*)&ptL[r][s] = ll;
  }
  __syncthreads();

  // ---- stage v transposed (vt[e][s]) over dead sc region ----
  __bf16 (*vtH)[VT_STR] = (__bf16(*)[VT_STR])(smem + OFF_VT_H);
  __bf16 (*vtL)[VT_STR] = (__bf16(*)[VT_STR])(smem + OFF_VT_L);
#pragma unroll
  for (int i = 0; i < 8; ++i) {
    int idx = tid + i * 256;               // 2048 units: e x s4
    int e = idx & 63, s4 = (idx >> 6) * 4; // consecutive lanes -> consecutive e
    float f0 = vbuf[rowbase + (size_t)(s4 + 0) * DQK + e];
    float f1 = vbuf[rowbase + (size_t)(s4 + 1) * DQK + e];
    float f2 = vbuf[rowbase + (size_t)(s4 + 2) * DQK + e];
    float f3 = vbuf[rowbase + (size_t)(s4 + 3) * DQK + e];
    split4_store(f0, f1, f2, f3, &vtH[e][s4], &vtL[e][s4]);
  }
  __syncthreads();

  // ---- O = P @ V  (M=128, N=64, K=128) ----
  {
    const int wm = wave & 3, wn = wave >> 2;  // 4(M) x 2(N) wave grid
    v8f oacc[2][2] = {};
#pragma unroll
    for (int ks = 0; ks < 128; ks += 32) {
#pragma unroll
      for (int mt = 0; mt < 2; ++mt) {
        const int ar = wm * 32 + mt * 16 + (lane & 15);
        v16bf ah = load_a_frag(&ptH[ar][ks], lane);
        v16bf al = load_a_frag(&ptL[ar][ks], lane);
#pragma unroll
        for (int nt = 0; nt < 2; ++nt) {
          const int br = wn * 32 + nt * 16 + (lane & 15);
          v16bf bh = load_b_frag(&vtH[br][ks], lane);
          v16bf bl = load_b_frag(&vtL[br][ks], lane);
          oacc[mt][nt] = wmma_comp(ah, al, bh, bl, oacc[mt][nt]);
        }
      }
    }
#pragma unroll
    for (int mt = 0; mt < 2; ++mt)
#pragma unroll
      for (int nt = 0; nt < 2; ++nt) {
        const int n = wn * 32 + nt * 16 + (lane & 15);
        const int mbase = wm * 32 + mt * 16 + ((lane & 16) ? 8 : 0);
#pragma unroll
        for (int r = 0; r < 8; ++r)
          ctx[rowbase + (size_t)(mbase + r) * DQK + n] = oacc[mt][nt][r];
      }
  }
}

// ---------------------------------------------------------------------------
// launch
// ---------------------------------------------------------------------------
extern "C" void kernel_launch(void* const* d_in, const int* in_sizes, int n_in,
                              void* d_out, int out_size, void* d_ws, size_t ws_size,
                              hipStream_t stream) {
  (void)in_sizes; (void)n_in; (void)out_size; (void)ws_size;

  const float* queries = (const float*)d_in[0];
  const float* keys    = (const float*)d_in[1];
  const float* values  = (const float*)d_in[2];
  const unsigned char* mask = (const unsigned char*)d_in[3];
  const float* Wq = (const float*)d_in[4];
  const float* bq = (const float*)d_in[5];
  const float* Wk = (const float*)d_in[6];
  const float* bk = (const float*)d_in[7];
  const float* Wv = (const float*)d_in[8];
  const float* bv = (const float*)d_in[9];
  const float* Wo = (const float*)d_in[10];
  const float* bo = (const float*)d_in[11];
  const float* pin  = (const float*)d_in[12];
  const float* pseq = (const float*)d_in[13];

  float* out  = (float*)d_out;                       // [B*L, D_MODEL]
  float* attn = out + (size_t)BL * DM;               // [B,H,L,S]

  // workspace: q, k, v, ctx  (4 * 32 MiB fp32)
  float* qb  = (float*)d_ws;
  float* kb  = qb + (size_t)BL * DQK;
  float* vb  = kb + (size_t)BL * DQK;
  float* ctx = vb + (size_t)BL * DQK;

  dim3 gproj(DQK / 128, BL / 128);   // (8, 64)
  gemm_bias_kernel<<<gproj, 256, 0, stream>>>(queries, Wq, bq, qb, BL, DQK, DM);
  gemm_bias_kernel<<<gproj, 256, 0, stream>>>(keys,    Wk, bk, kb, BL, DQK, DM);
  gemm_bias_kernel<<<gproj, 256, 0, stream>>>(values,  Wv, bv, vb, BL, DQK, DM);

  rope_kernel<<<(BL * 512) / 256, 256, 0, stream>>>(qb, kb, pin, pseq);

  attn_kernel<<<NB * NH, 256, 0, stream>>>(qb, kb, vb, mask, attn, ctx);

  dim3 gout(DM / 128, BL / 128);
  gemm_bias_kernel<<<gout, 256, 0, stream>>>(ctx, Wo, bo, out, BL, DM, DQK);
}
